// MultiHeadAttention_35871566856639
// MI455X (gfx1250) — matrix operand order repaired and checked by reading, not compiled, not run on hardware
//
#include <hip/hip_runtime.h>
#include <hip/hip_bf16.h>

typedef __attribute__((ext_vector_type(16))) __bf16 v16bf;
typedef __attribute__((ext_vector_type(8)))  float  v8f;

#define N_HEADS  12
#define HIDDEN   768
#define HEAD_DIM 64
#define BATCH    4
#define SEQ      2048
#define M_TOTAL  (BATCH * SEQ)   // 8192
#define FRAG     512             // elements per 32x16 / 16x32 bf16 fragment (32 lanes x 16)

// ---------------------------------------------------------------------------
// Fragment-native layouts (per CDNA5 ISA):
//  A-frag (16x32, MxK): element (m, kl) -> lane = 16*((kl>>3)&1) + m,
//                                          e    = ((kl>>4)<<3) | (kl&7)
//  B-frag (32x16, KxN): element (kl, n) -> lane = 16*((kl>>3)&1) + n,
//                                          e    = ((kl>>4)<<3) | (kl&7)
//  Each lane owns 16 contiguous bf16 (32 B) => fragment loads are 2x b128.
//  C/D (16x16 f32): lane holds col n=lane&15, vgpr i holds row m=i+8*(lane>>4)
// ---------------------------------------------------------------------------

#define WMMA(A, B, C) __builtin_amdgcn_wmma_f32_16x16x32_bf16(false, (A), false, (B), (short)0, (C), false, false)

// ---------------------------------------------------------------------------
// Pack X (fp32 [8192,768]) -> A-fragment-native bf16: Xf[m16][kk][lane][16]
// ---------------------------------------------------------------------------
__global__ void pack_x_kernel(const float* __restrict__ X, __bf16* __restrict__ Xf) {
  int idx = blockIdx.x * 256 + threadIdx.x;       // coalesced read
  int m = idx / HIDDEN, k = idx % HIDDEN;
  int m16 = m >> 4, kk = k >> 5, kl = k & 31;
  int lane = (((kl >> 3) & 1) << 4) | (m & 15);
  int e = ((kl >> 4) << 3) | (kl & 7);
  Xf[((size_t)(m16 * 24 + kk)) * FRAG + lane * 16 + e] = (__bf16)X[idx];
}

// ---------------------------------------------------------------------------
// Pack W (fp32 [n][k] row-major, torch Linear) -> B-fragment-native bf16 for
// Y = X @ W^T:  B[k][n] = W[n][k].  Wf[kk][n16][lane][16]
// ---------------------------------------------------------------------------
__global__ void pack_w_kernel(const float* __restrict__ W, __bf16* __restrict__ Wf) {
  int idx = blockIdx.x * 256 + threadIdx.x;       // coalesced read along k
  int n = idx / HIDDEN, k = idx % HIDDEN;
  int kk = k >> 5, kl = k & 31;
  int lane = (((kl >> 3) & 1) << 4) | (n & 15);
  int e = ((kl >> 4) << 3) | (kl & 7);
  Wf[((size_t)(kk * 48 + (n >> 4))) * FRAG + lane * 16 + e] = (__bf16)W[idx];
}

// ---------------------------------------------------------------------------
// QKV projection GEMM, fragment-native, LDS-free, pointer-increment k-loop:
// M=8192, N=768, K=768. Block=128 (4 waves); 64x64 tile; wave does 32x32.
// Epilogue scatters (with bias; Q also folds softmax scale) into
// attention-ready fragment layouts:
//   mode 0: Q  -> A-frag  Qf[bh][s16][kk2][512]           (k-dim = d)
//   mode 1: K  -> B-frag  Kf[bh][j16][kk2][512]           (k-dim = d, n = j)
//   mode 2: V  -> B-frag  Vf[bh][kkj][d16][512]           (k-dim = j, n = d)
// ---------------------------------------------------------------------------
__global__ __launch_bounds__(128) void qkv_gemm_kernel(
    const __bf16* __restrict__ Xf, const __bf16* __restrict__ Wf,
    const float* __restrict__ bias, __bf16* __restrict__ Out, int mode) {
  const int t = threadIdx.x;
  const int w = t >> 5;
  const int lane = t & 31;
  const int half = lane >> 4;
  const int cit  = lane & 15;
  const int wm = (w & 1) * 32;
  const int wn = (w >> 1) * 32;
  const int m0 = blockIdx.x * 64;
  const int n0 = blockIdx.y * 64;
  const int m16b = blockIdx.x * 4 + (wm >> 4);
  const int n16b = blockIdx.y * 4 + (wn >> 4);

  // Fragment base pointers; consecutive kk strides are constant:
  //   A: +FRAG elements (1 KB), B: +48*FRAG elements (48 KB).
  const __bf16* pa0 = Xf + ((size_t)(m16b + 0) * 24) * FRAG + lane * 16;
  const __bf16* pa1 = Xf + ((size_t)(m16b + 1) * 24) * FRAG + lane * 16;
  const __bf16* pb0 = Wf + ((size_t)(n16b + 0)) * FRAG + lane * 16;
  const __bf16* pb1 = Wf + ((size_t)(n16b + 1)) * FRAG + lane * 16;

  v8f acc[2][2] = {};

#pragma unroll 2
  for (int kk = 0; kk < 24; ++kk) {
    v16bf a0 = *(const v16bf*)pa0;
    v16bf a1 = *(const v16bf*)pa1;
    v16bf b0 = *(const v16bf*)pb0;
    v16bf b1 = *(const v16bf*)pb1;
    __builtin_prefetch(pa0 + FRAG, 0, 0);   // global_prefetch_b8, next k-step
    acc[0][0] = WMMA(a0, b0, acc[0][0]);
    acc[0][1] = WMMA(a0, b1, acc[0][1]);
    acc[1][0] = WMMA(a1, b0, acc[1][0]);
    acc[1][1] = WMMA(a1, b1, acc[1][1]);
    pa0 += FRAG;
    pa1 += FRAG;
    pb0 += 48 * FRAG;
    pb1 += 48 * FRAG;
  }

#pragma unroll
  for (int a = 0; a < 2; ++a)
#pragma unroll
    for (int bb = 0; bb < 2; ++bb)
#pragma unroll
      for (int i = 0; i < 8; ++i) {
        int m = m0 + wm + a * 16 + i + 8 * half;
        int n = n0 + wn + bb * 16 + cit;
        float val = acc[a][bb][i] + bias[n];
        if (mode == 0) val *= 0.125f;       // fold 1/sqrt(HEAD_DIM) into Q
        __bf16 ov = (__bf16)val;
        int b = m >> 11, s = m & (SEQ - 1);
        int h = n >> 6, d = n & 63;
        int bh = b * N_HEADS + h;
        size_t idx;
        if (mode == 0) {            // Q: A-frag over d
          int dl = d & 31;
          int lp = (((dl >> 3) & 1) << 4) | (s & 15);
          int e  = ((dl >> 4) << 3) | (dl & 7);
          idx = ((size_t)((bh * 128 + (s >> 4)) * 2 + (d >> 5))) * FRAG + lp * 16 + e;
        } else if (mode == 1) {     // K: B-frag, k=d, n=j(=s)
          int dl = d & 31;
          int lp = (((dl >> 3) & 1) << 4) | (s & 15);
          int e  = ((dl >> 4) << 3) | (dl & 7);
          idx = ((size_t)((bh * 128 + (s >> 4)) * 2 + (d >> 5))) * FRAG + lp * 16 + e;
        } else {                    // V: B-frag, k=j(=s), n=d
          int sl = s & 31;
          int lp = (((sl >> 3) & 1) << 4) | (d & 15);
          int e  = ((sl >> 4) << 3) | (sl & 7);
          idx = ((size_t)((bh * 64 + (s >> 5)) * 4 + (d >> 4))) * FRAG + lp * 16 + e;
        }
        Out[idx] = ov;
      }
}

// ---------------------------------------------------------------------------
// Flash attention. Block = 128 threads (4 waves) per (b*h, 64-query tile);
// each wave owns 16 query rows. K/V tiles (8 KB each, contiguous fragment
// blocks) staged via global_load_async_to_lds_b128 (ASYNCcnt). Online softmax
// in-wave via shfl_xor; P round-trips per-wave LDS (C-layout -> A-frag).
// ---------------------------------------------------------------------------
__global__ __launch_bounds__(128) void attn_kernel(
    const __bf16* __restrict__ Qf, const __bf16* __restrict__ Kf,
    const __bf16* __restrict__ Vf, float* __restrict__ Out) {
  __shared__ __align__(32) __bf16 Ks[8 * FRAG];   // frags [j16sub][kk2]
  __shared__ __align__(32) __bf16 Vs[8 * FRAG];   // frags [kkjsub][d16]
  __shared__ __align__(32) __bf16 Ps[4 * 1024];   // per-wave P A-frags [kkp][512]

  const int t = threadIdx.x;
  const int w = t >> 5;
  const int lane = t & 31;
  const int half = lane >> 4;
  const int cit  = lane & 15;
  const int bh = blockIdx.y;
  const int q0 = blockIdx.x * 64;

  // Per-wave Q A-fragments, straight from global (2x b128 each). Q is
  // pre-scaled by 1/sqrt(HEAD_DIM) in the projection epilogue.
  const int s16 = (q0 >> 4) + w;
  v16bf qa0 = *(const v16bf*)(Qf + ((size_t)((bh * 128 + s16) * 2 + 0)) * FRAG + lane * 16);
  v16bf qa1 = *(const v16bf*)(Qf + ((size_t)((bh * 128 + s16) * 2 + 1)) * FRAG + lane * 16);

  v8f o[4] = {};
  float rmax[8], rsum[8];
#pragma unroll
  for (int i = 0; i < 8; ++i) { rmax[i] = -1e30f; rsum[i] = 0.0f; }

  const unsigned ks_base = (unsigned)(size_t)&Ks[0];
  const unsigned vs_base = (unsigned)(size_t)&Vs[0];
  __bf16* Pw = &Ps[w * 1024];

  // Tile base pointers; each 64-key step advances both by 8 KB (4096 elems).
  const __bf16* kt_g = Kf + ((size_t)bh * 128) * 2 * FRAG;
  const __bf16* vt_g = Vf + ((size_t)bh * 64) * 4 * FRAG;

  for (int j0 = 0; j0 < SEQ; j0 += 64) {
    __syncthreads();
    // Async-stage K/V tiles: 8 KB contiguous each; 4x b128 per thread per tile.
#pragma unroll
    for (int i = 0; i < 4; ++i) {
      unsigned off = (unsigned)(t + i * 128) * 16u;   // 0..8191 bytes
      asm volatile("global_load_async_to_lds_b128 %0, %1, %2"
                   :: "v"(ks_base + off), "v"(off), "s"(kt_g) : "memory");
      asm volatile("global_load_async_to_lds_b128 %0, %1, %2"
                   :: "v"(vs_base + off), "v"(off), "s"(vt_g) : "memory");
    }
    asm volatile("s_wait_asynccnt 0x0" ::: "memory");
    __syncthreads();
    kt_g += 8 * FRAG;
    vt_g += 8 * FRAG;

    // S = Q @ K^T (pre-scaled): 4 j-subtiles x 2 k-steps over d.
    v8f sacc[4] = {};
#pragma unroll
    for (int ns = 0; ns < 4; ++ns) {
      v16bf b0 = *(const v16bf*)(Ks + (ns * 2 + 0) * FRAG + lane * 16);
      sacc[ns] = WMMA(qa0, b0, sacc[ns]);
      v16bf b1 = *(const v16bf*)(Ks + (ns * 2 + 1) * FRAG + lane * 16);
      sacc[ns] = WMMA(qa1, b1, sacc[ns]);
    }

    // Online softmax; each row lives in one 16-lane half.
#pragma unroll
    for (int i = 0; i < 8; ++i) {
      float tmax = fmaxf(fmaxf(sacc[0][i], sacc[1][i]), fmaxf(sacc[2][i], sacc[3][i]));
#pragma unroll
      for (int msk = 1; msk <= 8; msk <<= 1)
        tmax = fmaxf(tmax, __shfl_xor(tmax, msk, 32));
      float nm = fmaxf(rmax[i], tmax);
      float corr = __expf(rmax[i] - nm);
      rmax[i] = nm;
      float ps = 0.0f;
#pragma unroll
      for (int ns = 0; ns < 4; ++ns) {
        float p = __expf(sacc[ns][i] - nm);
        sacc[ns][i] = p;
        ps += p;
      }
#pragma unroll
      for (int msk = 1; msk <= 8; msk <<= 1)
        ps += __shfl_xor(ps, msk, 32);
      rsum[i] = rsum[i] * corr + ps;
#pragma unroll
      for (int ns = 0; ns < 4; ++ns) o[ns][i] *= corr;
    }

    // P: C-layout -> A-frag layout in per-wave LDS scratch.
#pragma unroll
    for (int ns = 0; ns < 4; ++ns)
#pragma unroll
      for (int i = 0; i < 8; ++i) {
        int lp = (((cit >> 3) & 1) << 4) | (i + 8 * half);
        int e  = ((ns & 1) << 3) | (cit & 7);
        Pw[(ns >> 1) * FRAG + lp * 16 + e] = (__bf16)sacc[ns][i];
      }
    asm volatile("s_wait_dscnt 0x0" ::: "memory");
    v16bf pa0 = *(const v16bf*)(Pw + 0 * FRAG + lane * 16);
    v16bf pa1 = *(const v16bf*)(Pw + 1 * FRAG + lane * 16);

    // O += P @ V: 4 d-subtiles x 2 k-steps over j.
#pragma unroll
    for (int d16 = 0; d16 < 4; ++d16) {
      v16bf vb0 = *(const v16bf*)(Vs + (0 * 4 + d16) * FRAG + lane * 16);
      o[d16] = WMMA(pa0, vb0, o[d16]);
      v16bf vb1 = *(const v16bf*)(Vs + (1 * 4 + d16) * FRAG + lane * 16);
      o[d16] = WMMA(pa1, vb1, o[d16]);
    }
  }

  // Normalize, scatter to [B, S, HIDDEN] fp32.
  const int b = bh / N_HEADS;
  const int h = bh % N_HEADS;
#pragma unroll
  for (int d16 = 0; d16 < 4; ++d16)
#pragma unroll
    for (int i = 0; i < 8; ++i) {
      int m = q0 + w * 16 + i + 8 * half;
      int d = d16 * 16 + cit;
      Out[((size_t)b * SEQ + m) * HIDDEN + h * HEAD_DIM + d] = o[d16][i] / rsum[i];
    }
}

// ---------------------------------------------------------------------------
extern "C" void kernel_launch(void* const* d_in, const int* in_sizes, int n_in,
                              void* d_out, int out_size, void* d_ws, size_t ws_size,
                              hipStream_t stream) {
  (void)in_sizes; (void)n_in; (void)out_size; (void)ws_size;
  const float* X  = (const float*)d_in[0];
  const float* Wq = (const float*)d_in[1];
  const float* bq = (const float*)d_in[2];
  const float* Wk = (const float*)d_in[3];
  const float* bk = (const float*)d_in[4];
  const float* Wv = (const float*)d_in[5];
  const float* bv = (const float*)d_in[6];
  float* out = (float*)d_out;

  // Workspace (bf16): Xf | Wfq | Wfk | Wfv | Qf | Kf | Vf   (~54 MB)
  __bf16* Xf  = (__bf16*)d_ws;
  __bf16* Wfq = Xf  + (size_t)M_TOTAL * HIDDEN;
  __bf16* Wfk = Wfq + (size_t)HIDDEN * HIDDEN;
  __bf16* Wfv = Wfk + (size_t)HIDDEN * HIDDEN;
  __bf16* Qf  = Wfv + (size_t)HIDDEN * HIDDEN;
  __bf16* Kf  = Qf  + (size_t)M_TOTAL * HIDDEN;
  __bf16* Vf  = Kf  + (size_t)M_TOTAL * HIDDEN;

  pack_x_kernel<<<(M_TOTAL * HIDDEN) / 256, 256, 0, stream>>>(X, Xf);
  pack_w_kernel<<<(HIDDEN * HIDDEN) / 256, 256, 0, stream>>>(Wq, Wfq);
  pack_w_kernel<<<(HIDDEN * HIDDEN) / 256, 256, 0, stream>>>(Wk, Wfk);
  pack_w_kernel<<<(HIDDEN * HIDDEN) / 256, 256, 0, stream>>>(Wv, Wfv);

  dim3 gg(M_TOTAL / 64, HIDDEN / 64);
  qkv_gemm_kernel<<<gg, 128, 0, stream>>>(Xf, Wfq, bq, Qf, 0);
  qkv_gemm_kernel<<<gg, 128, 0, stream>>>(Xf, Wfk, bk, Kf, 1);
  qkv_gemm_kernel<<<gg, 128, 0, stream>>>(Xf, Wfv, bv, Vf, 2);

  attn_kernel<<<dim3(SEQ / 64, BATCH * N_HEADS), 128, 0, stream>>>(Qf, Kf, Vf, out);
}
